// TextGenerationModel_2911987827449
// MI455X (gfx1250) — matrix operand order repaired and checked by reading, not compiled, not run on hardware
//
#include <hip/hip_runtime.h>

// Problem constants (match reference)
#define B_ 128
#define T_ 256
#define V_ 256
#define H_ 1024
#define L_ 2
#define KC 1280   // H + V : fused K for [h | x] gate GEMM
#define NG 4096   // 4*H   : 4 gates side by side

typedef __attribute__((ext_vector_type(8)))  float  v8f;
typedef __attribute__((ext_vector_type(8)))  __bf16 v8bf;
typedef __attribute__((ext_vector_type(16))) __bf16 v16bf;

static __device__ __forceinline__ v16bf cat8(v8bf lo, v8bf hi) {
    return __builtin_shufflevector(lo, hi, 0,1,2,3,4,5,6,7,8,9,10,11,12,13,14,15);
}
static __device__ __forceinline__ float sigf(float x)   { return 1.0f / (1.0f + __expf(-x)); }
static __device__ __forceinline__ float tanh_f(float x) { return 1.0f - 2.0f / (__expf(2.0f * x) + 1.0f); }

// ---------------------------------------------------------------------------
// Weight prep: Wt[l][n=g*H+col][k] (bf16, K-contiguous) with k<H from W_{g}h,
// k>=H from W_{g}x.  Each WMMA B-fragment lane reads two 8-element
// K runs, k+8h and k+16+8h (matches the 16-bit 32x16 B layout).
// ---------------------------------------------------------------------------
__global__ void convert_wcat(const float* __restrict__ Wgh, const float* __restrict__ Wih,
                             const float* __restrict__ Wfh, const float* __restrict__ Woh,
                             const float* __restrict__ Wgx, const float* __restrict__ Wix,
                             const float* __restrict__ Wfx, const float* __restrict__ Wox,
                             __bf16* __restrict__ Wt) {
    long g = (long)blockIdx.x * blockDim.x + threadIdx.x;
    const long total = (long)L_ * NG * KC;
    if (g >= total) return;
    int  k    = (int)(g % KC);
    long r    = g / KC;
    int  n    = (int)(r % NG);
    int  l    = (int)(r / NG);
    int  gate = n >> 10;
    int  ncol = n & (H_ - 1);
    const float* Wh = gate == 0 ? Wgh : gate == 1 ? Wih : gate == 2 ? Wfh : Woh;
    const float* Wx = gate == 0 ? Wgx : gate == 1 ? Wix : gate == 2 ? Wfx : Wox;
    float v;
    if (k < H_) v = Wh[((long)l * H_ + k) * H_ + ncol];
    else        v = Wx[((long)l * V_ + (k - H_)) * H_ + ncol];
    Wt[g] = (__bf16)v;
}

// Wpht[l][v][h] = W_ph[l][h][v]  (bf16, K(=h)-contiguous per output column v)
__global__ void convert_wph(const float* __restrict__ Wph, __bf16* __restrict__ Wpht) {
    long g = (long)blockIdx.x * blockDim.x + threadIdx.x;
    const long total = (long)L_ * V_ * H_;
    if (g >= total) return;
    int  h = (int)(g % H_);
    long r = g / H_;
    int  v = (int)(r % V_);
    int  l = (int)(r / V_);
    Wpht[g] = (__bf16)Wph[((long)l * H_ + h) * V_ + v];
}

// One-hot tokens, time-major, bf16: Xin0[t][b][v] = (x[b][t]==v)
__global__ void onehot_embed(const int* __restrict__ x, __bf16* __restrict__ Xin0) {
    long g = (long)blockIdx.x * blockDim.x + threadIdx.x;
    const long total = (long)T_ * B_ * V_;
    if (g >= total) return;
    int v = (int)(g & (V_ - 1));
    int b = (int)((g >> 8) & (B_ - 1));
    int t = (int)(g >> 15);
    Xin0[g] = (__bf16)((x[b * T_ + t] == v) ? 1.0f : 0.0f);
}

__global__ void zero_state(__bf16* __restrict__ Hbuf, float* __restrict__ Cbuf) {
    int i = blockIdx.x * blockDim.x + threadIdx.x;   // grid covers B_*H_
    Cbuf[i] = 0.0f;
    Hbuf[i] = (__bf16)0.0f;
    Hbuf[i + B_ * H_] = (__bf16)0.0f;
}

// ---------------------------------------------------------------------------
// Kernel 1: one LSTM timestep.  Grid = 64 WGs (4 batch-tiles x 16 h-slices),
// 256 threads = 8 waves.  Wave (mt,gate) computes a 16x64 slab of gate 'gate'
// pre-activations with a software-pipelined K loop (fragments for chunk k+1
// are loaded while the 4 WMMAs for chunk k execute), then gates are exchanged
// through LDS and the c/h update finishes locally.
// ---------------------------------------------------------------------------
__global__ void __launch_bounds__(256)
lstm_step(const __bf16* __restrict__ Wt,     // [NG][KC] bf16
          const __bf16* __restrict__ Xin_t,  // [B_][V_] bf16 (this timestep's input)
          __bf16* __restrict__ Hbuf,         // [2][B_][H_] ping-pong
          float*  __restrict__ Cbuf,         // [B_][H_]
          const float* __restrict__ bg, const float* __restrict__ bi,
          const float* __restrict__ bff, const float* __restrict__ bo, int t)
{
    __shared__ float s_pre[2][4][16][64];   // 32 KB

    const int tid   = threadIdx.x;
    const int wave  = tid >> 5, lane = tid & 31;
    const int mt    = wave >> 2, gate = wave & 3;
    const int lane15 = lane & 15, hi = lane >> 4;
    const int rowbase = (blockIdx.x & 3) * 32 + mt * 16;
    const int colbase = (blockIdx.x >> 2) * 64;

    const __bf16* Hprev = Hbuf + (size_t)(t & 1) * (B_ * H_);
    const int arow_idx = rowbase + lane15;

    // Per-lane row bases (the +hi*8 swizzle folded in once).
    const __bf16* hrow = Hprev + (size_t)arow_idx * H_ + (hi << 3);
    const __bf16* xrow = Xin_t + (size_t)arow_idx * V_ + (hi << 3);
    const __bf16* wcol = Wt + (size_t)(gate * H_ + colbase + lane15) * KC + (hi << 3);

    auto loadA = [&](int k) -> v16bf {
        const __bf16* p = (k < H_) ? (hrow + k) : (xrow + (k - H_));
        return cat8(*(const v8bf*)p, *(const v8bf*)(p + 16));
    };
    auto loadB = [&](int k, int nt) -> v16bf {
        const __bf16* p = wcol + (size_t)nt * (16 * KC) + k;
        return cat8(*(const v8bf*)p, *(const v8bf*)(p + 16));
    };

    v8f acc[4] = {{}, {}, {}, {}};

    // Software pipeline: prologue loads chunk 0; each iteration loads chunk
    // k+1 into fresh registers before issuing the WMMAs for chunk k.
    v16bf aC  = loadA(0);
    v16bf bC0 = loadB(0, 0), bC1 = loadB(0, 1), bC2 = loadB(0, 2), bC3 = loadB(0, 3);

    for (int k0 = 0; k0 < KC - 32; k0 += 32) {
        const int k1 = k0 + 32;
        v16bf aN  = loadA(k1);
        v16bf bN0 = loadB(k1, 0), bN1 = loadB(k1, 1);
        v16bf bN2 = loadB(k1, 2), bN3 = loadB(k1, 3);

        acc[0] = __builtin_amdgcn_wmma_f32_16x16x32_bf16(false, aC, false, bC0, (short)0, acc[0], false, false);
        acc[1] = __builtin_amdgcn_wmma_f32_16x16x32_bf16(false, aC, false, bC1, (short)0, acc[1], false, false);
        acc[2] = __builtin_amdgcn_wmma_f32_16x16x32_bf16(false, aC, false, bC2, (short)0, acc[2], false, false);
        acc[3] = __builtin_amdgcn_wmma_f32_16x16x32_bf16(false, aC, false, bC3, (short)0, acc[3], false, false);

        aC = aN; bC0 = bN0; bC1 = bN1; bC2 = bN2; bC3 = bN3;
    }
    // Epilogue: last chunk.
    acc[0] = __builtin_amdgcn_wmma_f32_16x16x32_bf16(false, aC, false, bC0, (short)0, acc[0], false, false);
    acc[1] = __builtin_amdgcn_wmma_f32_16x16x32_bf16(false, aC, false, bC1, (short)0, acc[1], false, false);
    acc[2] = __builtin_amdgcn_wmma_f32_16x16x32_bf16(false, aC, false, bC2, (short)0, acc[2], false, false);
    acc[3] = __builtin_amdgcn_wmma_f32_16x16x32_bf16(false, aC, false, bC3, (short)0, acc[3], false, false);

    // bias + park pre-activations in LDS (C/D layout: VGPR j -> M=j+hi*8, N=lane15)
    const float* bgate = (gate == 0 ? bg : gate == 1 ? bi : gate == 2 ? bff : bo);
    #pragma unroll
    for (int nt = 0; nt < 4; ++nt) {
        int cloc = nt * 16 + lane15;
        float bias = bgate[colbase + cloc];
        #pragma unroll
        for (int j = 0; j < 8; ++j)
            s_pre[mt][gate][j + hi * 8][cloc] = acc[nt][j] + bias;
    }
    __syncthreads();

    // c/h update: 32 rows x 64 cols, 8 elements per thread
    __bf16* Hnext = Hbuf + (size_t)((t + 1) & 1) * (B_ * H_);
    int r   = tid >> 3;
    int cb  = (tid & 7) * 8;
    int mt2 = r >> 4, rl = r & 15;
    int growb = (blockIdx.x & 3) * 32 + r;
    #pragma unroll
    for (int i = 0; i < 8; ++i) {
        int cl = cb + i;
        int gcol = colbase + cl;
        float gg = tanh_f(s_pre[mt2][0][rl][cl]);
        float ii = sigf  (s_pre[mt2][1][rl][cl]);
        float ff = sigf  (s_pre[mt2][2][rl][cl]);
        float oo = sigf  (s_pre[mt2][3][rl][cl]);
        size_t idx = (size_t)growb * H_ + gcol;
        float cnew = gg * ii + Cbuf[idx] * ff;
        Cbuf[idx]  = cnew;
        Hnext[idx] = (__bf16)(tanh_f(cnew) * oo);
    }
}

// ---------------------------------------------------------------------------
// Kernel 2: y = softmax(h @ Wph + bp).  Grid = 8 WGs (one 16-row batch tile),
// 8 waves x 2 N-tiles cover the 256 vocab columns; pipelined WMMA GEMM then
// LDS softmax.  Layer 0 writes bf16 into layer 1's input stream, layer 1
// writes fp32 d_out.
// ---------------------------------------------------------------------------
__global__ void __launch_bounds__(256)
lstm_out(const __bf16* __restrict__ Wpht,   // [V_][H_] bf16
         const __bf16* __restrict__ Hcur,   // [B_][H_] bf16 (new h)
         const float* __restrict__ bp, void* __restrict__ outp, int out_f32)
{
    __shared__ float s_log[16][256];        // 16 KB
    __shared__ float s_mx[16];
    __shared__ float s_inv[16];

    const int tid = threadIdx.x, wave = tid >> 5, lane = tid & 31;
    const int lane15 = lane & 15, hi = lane >> 4;
    const int rowbase = blockIdx.x * 16;

    const __bf16* arow = Hcur + (size_t)(rowbase + lane15) * H_ + (hi << 3);
    const __bf16* wb   = Wpht + (size_t)(wave * 32 + lane15) * H_ + (hi << 3);

    auto loadA = [&](int k) -> v16bf {
        return cat8(*(const v8bf*)(arow + k), *(const v8bf*)(arow + k + 16));
    };
    auto loadB = [&](int k, int nt) -> v16bf {
        const __bf16* p = wb + (size_t)nt * (16 * H_) + k;
        return cat8(*(const v8bf*)p, *(const v8bf*)(p + 16));
    };

    v8f acc[2] = {{}, {}};
    v16bf aC  = loadA(0);
    v16bf bC0 = loadB(0, 0), bC1 = loadB(0, 1);

    for (int k0 = 0; k0 < H_ - 32; k0 += 32) {
        const int k1 = k0 + 32;
        v16bf aN  = loadA(k1);
        v16bf bN0 = loadB(k1, 0), bN1 = loadB(k1, 1);
        acc[0] = __builtin_amdgcn_wmma_f32_16x16x32_bf16(false, aC, false, bC0, (short)0, acc[0], false, false);
        acc[1] = __builtin_amdgcn_wmma_f32_16x16x32_bf16(false, aC, false, bC1, (short)0, acc[1], false, false);
        aC = aN; bC0 = bN0; bC1 = bN1;
    }
    acc[0] = __builtin_amdgcn_wmma_f32_16x16x32_bf16(false, aC, false, bC0, (short)0, acc[0], false, false);
    acc[1] = __builtin_amdgcn_wmma_f32_16x16x32_bf16(false, aC, false, bC1, (short)0, acc[1], false, false);

    #pragma unroll
    for (int nt = 0; nt < 2; ++nt) {
        int col = wave * 32 + nt * 16 + lane15;
        float bias = bp[col];
        #pragma unroll
        for (int j = 0; j < 8; ++j)
            s_log[j + hi * 8][col] = acc[nt][j] + bias;
    }
    __syncthreads();

    if (tid < 16) {
        float m = -3.4e38f;
        for (int c = 0; c < V_; ++c) m = fmaxf(m, s_log[tid][c]);
        float s = 0.0f;
        for (int c = 0; c < V_; ++c) s += __expf(s_log[tid][c] - m);
        s_mx[tid]  = m;
        s_inv[tid] = 1.0f / s;
    }
    __syncthreads();

    int r = tid >> 4, cb2 = (tid & 15) * 16;
    float m = s_mx[r], inv = s_inv[r];
    size_t obase = (size_t)(rowbase + r) * V_;
    for (int i = 0; i < 16; ++i) {
        int c = cb2 + i;
        float y = __expf(s_log[r][c] - m) * inv;
        if (out_f32) ((float*) outp)[obase + c] = y;
        else         ((__bf16*)outp)[obase + c] = (__bf16)y;
    }
}

// ---------------------------------------------------------------------------
extern "C" void kernel_launch(void* const* d_in, const int* in_sizes, int n_in,
                              void* d_out, int out_size, void* d_ws, size_t ws_size,
                              hipStream_t stream)
{
    const int*   x   = (const int*)  d_in[0];
    const float* Wgx = (const float*)d_in[1];
    const float* Wix = (const float*)d_in[2];
    const float* Wfx = (const float*)d_in[3];
    const float* Wox = (const float*)d_in[4];
    const float* Wgh = (const float*)d_in[5];
    const float* Wih = (const float*)d_in[6];
    const float* Wfh = (const float*)d_in[7];
    const float* Woh = (const float*)d_in[8];
    const float* Wph = (const float*)d_in[9];
    const float* b_g = (const float*)d_in[10];
    const float* b_i = (const float*)d_in[11];
    const float* b_f = (const float*)d_in[12];
    const float* b_o = (const float*)d_in[13];
    const float* b_p = (const float*)d_in[14];

    char* ws = (char*)d_ws;
    size_t off = 0;
    auto take = [&](size_t bytes) -> char* {
        char* p = ws + off;
        off = (off + bytes + 255) & ~(size_t)255;
        return p;
    };
    __bf16* Wt   = (__bf16*)take((size_t)L_ * NG * KC * 2);   // 21.0 MB
    __bf16* Wpht = (__bf16*)take((size_t)L_ * V_ * H_ * 2);   //  1.0 MB
    __bf16* Xin0 = (__bf16*)take((size_t)T_ * B_ * V_ * 2);   // 16.8 MB
    __bf16* Xin1 = (__bf16*)take((size_t)T_ * B_ * V_ * 2);   // 16.8 MB
    __bf16* Hbuf = (__bf16*)take((size_t)2 * B_ * H_ * 2);
    float*  Cbuf = (float*) take((size_t)B_ * H_ * 4);

    {
        long total = (long)L_ * NG * KC;
        convert_wcat<<<(int)((total + 255) / 256), 256, 0, stream>>>(
            Wgh, Wih, Wfh, Woh, Wgx, Wix, Wfx, Wox, Wt);
    }
    {
        long total = (long)L_ * V_ * H_;
        convert_wph<<<(int)((total + 255) / 256), 256, 0, stream>>>(Wph, Wpht);
    }
    {
        long total = (long)T_ * B_ * V_;
        onehot_embed<<<(int)((total + 255) / 256), 256, 0, stream>>>(x, Xin0);
    }

    for (int l = 0; l < L_; ++l) {
        zero_state<<<(B_ * H_) / 256, 256, 0, stream>>>(Hbuf, Cbuf);
        const __bf16* WtL   = Wt   + (size_t)l * NG * KC;
        const __bf16* WphtL = Wpht + (size_t)l * V_ * H_;
        const __bf16* XinL  = (l == 0) ? Xin0 : Xin1;
        const float *bg = b_g + l * H_, *bi = b_i + l * H_;
        const float *bf2 = b_f + l * H_, *bo = b_o + l * H_, *bpp = b_p + l * V_;
        for (int t = 0; t < T_; ++t) {
            lstm_step<<<64, 256, 0, stream>>>(WtL, XinL + (size_t)t * B_ * V_,
                                              Hbuf, Cbuf, bg, bi, bf2, bo, t);
            void* outT = (l == 0) ? (void*)(Xin1 + (size_t)t * B_ * V_)
                                  : (void*)((float*)d_out + (size_t)t * B_ * V_);
            lstm_out<<<8, 256, 0, stream>>>(WphtL, Hbuf + (size_t)((t + 1) & 1) * B_ * H_,
                                            bpp, outT, l);
        }
    }
}
